// Pathology_Balanced_Prompting_Module_19937238188507
// MI455X (gfx1250) — matrix operand order repaired and checked by reading, not compiled, not run on hardware
//
#include <hip/hip_runtime.h>

typedef __attribute__((ext_vector_type(16))) __bf16 v16bf;
typedef __attribute__((ext_vector_type(4)))  __bf16 v4bf;
typedef __attribute__((ext_vector_type(8)))  float  v8f;
typedef __attribute__((ext_vector_type(4)))  int    v4i;

#define BM 128
#define BN 128
#define BK 32
#define LDSS 40   // shorts per LDS row: 32 payload + 8 pad (16B-aligned rows, skewed banks)

#if defined(__HIP_DEVICE_COMPILE__) && \
    __has_builtin(__builtin_amdgcn_global_load_async_to_lds_b128) && \
    __has_builtin(__builtin_amdgcn_s_wait_asynccnt)
#define USE_ASYNC_LDS 1
#else
#define USE_ASYNC_LDS 0
#endif

__device__ __forceinline__ unsigned short bf16bits(float f) {
  union { __bf16 h; unsigned short s; } u;
  u.h = (__bf16)f;               // hardware RNE convert (v_cvt_pk_bf16_f32 class)
  return u.s;
}

__device__ __forceinline__ void packStoreA(unsigned short* dst, float4 v) {
  union { v4bf h; uint2 u; } p;
  p.h[0] = (__bf16)v.x; p.h[1] = (__bf16)v.y;
  p.h[2] = (__bf16)v.z; p.h[3] = (__bf16)v.w;
  *(uint2*)dst = p.u;            // ds_store_b64, 8B aligned
}

// Weight pre-pass: W[K][N] fp32 -> Wt[Npad][K32] bf16, transposed + zero-padded.
__global__ __launch_bounds__(256) void convert_wt_kernel(
    const float* __restrict__ W, unsigned short* __restrict__ Wt,
    int K, int N, int K32, int Npad)
{
  size_t gid = (size_t)blockIdx.x * 256 + threadIdx.x;
  size_t total = (size_t)K32 * Npad;
  if (gid >= total) return;
  int k = (int)(gid % K32);
  int n = (int)(gid / K32);
  float v = (k < K && n < N) ? W[(size_t)k * N + n] : 0.0f;
  Wt[gid] = bf16bits(v);
}

// C[M,N] = epi(A[M,K] @ W[K,N] + bias) * alpha + beta * res   (epi: 0=none,1=relu,2=sigmoid)
// A: fp32 row-major. Wt: bf16, transposed [Npad][K32], zero-padded (Npad >= gridDim.x*BN).
// Requires M % 128 == 0 (all call sites satisfy this).
__global__ __launch_bounds__(256) void gemm_wmma_kernel(
    const float* __restrict__ A, const unsigned short* __restrict__ Wt,
    const float* __restrict__ bias, const float* __restrict__ res,
    float* __restrict__ C, int M, int N, int K, int K32,
    float alpha, float beta, int epi)
{
  __shared__ unsigned short As[BM * LDSS];   // A tile bf16 [128][32+pad]
  __shared__ unsigned short Bs[BN * LDSS];   // B tile bf16 [n][k] [128][32+pad]

  const int tid  = threadIdx.x;
  const int lane = tid & 31;
  const int wv   = tid >> 5;        // 0..7 waves; wave wv owns rows 16*wv..16*wv+15
  const int r    = lane & 15;
  const int hi   = lane >> 4;

  const int n0 = blockIdx.x * BN;
  const int m0 = blockIdx.y * BM;
  const bool aVec = ((K & 3) == 0);

  // A staging registers: 128x32 fp32 = 1024 float4, 4 per thread
  float4 aReg[4];

  auto loadA = [&](int k0) {
    const bool aF = aVec && (k0 + BK <= K);
#pragma unroll
    for (int i = 0; i < 4; ++i) {
      int f = i * 256 + tid;
      int row = f >> 3, c4 = f & 7;             // row 0..127, 8 float4 per 32-float row
      if (aF) {
        aReg[i] = *(const float4*)(A + (size_t)(m0 + row) * K + k0 + c4 * 4);
      } else {
        float4 t = make_float4(0.f, 0.f, 0.f, 0.f);
        int gm = m0 + row;
        if (gm < M) {
          int kb = k0 + c4 * 4;
          const float* p = A + (size_t)gm * K + kb;
          if (kb + 0 < K) t.x = p[0];
          if (kb + 1 < K) t.y = p[1];
          if (kb + 2 < K) t.z = p[2];
          if (kb + 3 < K) t.w = p[3];
        }
        aReg[i] = t;
      }
    }
  };

  auto storeA = [&]() {
#pragma unroll
    for (int i = 0; i < 4; ++i) {
      int f = i * 256 + tid;
      int row = f >> 3, c4 = f & 7;
      packStoreA(&As[row * LDSS + c4 * 4], aReg[i]);
    }
  };

  v8f acc[8] = {};
  const int ksteps = (K + BK - 1) / BK;

  loadA(0);
  for (int kk = 0; kk < ksteps; ++kk) {
    const int k0 = kk * BK;
    // B tile: bf16 global -> LDS, 512 chunks of 16B, 2 per thread (rows always in-bounds: padded)
#pragma unroll
    for (int i = 0; i < 2; ++i) {
      int c = i * 256 + tid;
      int row = c >> 2, cc = c & 3;            // row = n 0..127, 4 x 16B per 32-bf16 row
      const unsigned short* src = Wt + (size_t)(n0 + row) * K32 + k0 + cc * 8;
      unsigned short* dst = &Bs[row * LDSS + cc * 8];
#if USE_ASYNC_LDS
      __builtin_amdgcn_global_load_async_to_lds_b128(
          (__attribute__((address_space(1))) v4i*)src,
          (__attribute__((address_space(3))) v4i*)dst, 0, 0);
#else
      *(uint4*)dst = *(const uint4*)src;
#endif
    }
    storeA();
    if (kk + 1 < ksteps) loadA((kk + 1) * BK);   // overlap next A global loads with WMMA
#if USE_ASYNC_LDS
    __builtin_amdgcn_s_wait_asynccnt(0);
#endif
    __syncthreads();

    union Frag { v16bf v; uint4 q[2]; };
    // A fragment (16x32 bf16): lane holds row r; K = hi*16 + 0..15
    Frag af;
    {
      const unsigned short* pa = &As[(wv * 16 + r) * LDSS + hi * 16];
      af.q[0] = *(const uint4*)pa;
      af.q[1] = *(const uint4*)(pa + 8);
    }
    // 8 B fragments (32x16 bf16): lane holds col c = cb*16+r; K = hi*16 + 0..15
    // Ping-pong fragment buffers so ds_load of cb+1 overlaps WMMA of cb.
    Frag bfr[2];
    {
      const unsigned short* pb = &Bs[(0 * 16 + r) * LDSS + hi * 16];
      bfr[0].q[0] = *(const uint4*)pb;
      bfr[0].q[1] = *(const uint4*)(pb + 8);
    }
#pragma unroll
    for (int cb = 0; cb < 8; ++cb) {
      const int cur = cb & 1;
      if (cb + 1 < 8) {
        const unsigned short* pb = &Bs[((cb + 1) * 16 + r) * LDSS + hi * 16];
        bfr[cur ^ 1].q[0] = *(const uint4*)pb;
        bfr[cur ^ 1].q[1] = *(const uint4*)(pb + 8);
      }
      acc[cb] = __builtin_amdgcn_wmma_f32_16x16x32_bf16(
          false, af.v, false, bfr[cur].v, (short)0, acc[cb], false, false);
    }
    __syncthreads();
  }

  // epilogue: D layout m = i + hi*8, n = r within each 16x16 fragment
#pragma unroll
  for (int cb = 0; cb < 8; ++cb) {
    int gn = n0 + cb * 16 + r;
    if (gn >= N) continue;
    float bv = bias ? bias[gn] : 0.0f;
#pragma unroll
    for (int i = 0; i < 8; ++i) {
      int gm = m0 + wv * 16 + hi * 8 + i;
      if (gm >= M) continue;
      float t = acc[cb][i] + bv;
      if (epi == 1)      t = fmaxf(t, 0.0f);
      else if (epi == 2) t = 1.0f / (1.0f + __expf(-t));
      float o = alpha * t;
      if (res) o += beta * res[(size_t)gm * N + gn];
      C[(size_t)gm * N + gn] = o;
    }
  }
}

// one thread per (b, h, q): scores over TK keys, softmax, weighted sum of V. H=8, DK=64, D=512.
template<int TQ, int TK, bool KNOW>
__global__ __launch_bounds__(256) void attn_kernel(
    const float* __restrict__ Q, const float* __restrict__ Km,
    const float* __restrict__ Vm, float* __restrict__ O,
    const float* __restrict__ mask, int Bn)
{
  int gid = blockIdx.x * blockDim.x + threadIdx.x;
  if (gid >= Bn * 8 * TQ) return;
  int q = gid % TQ;
  int h = (gid / TQ) % 8;
  int b = gid / (TQ * 8);

  const float* qp = Q + ((size_t)b * TQ + q) * 512 + h * 64;
  float4 qv[16];
#pragma unroll
  for (int i = 0; i < 16; ++i) qv[i] = *(const float4*)(qp + 4 * i);

  float s[TK];
  float mx = -3.0e38f;
#pragma unroll
  for (int k = 0; k < TK; ++k) {
    const float* kp = Km + ((size_t)b * TK + k) * 512 + h * 64;
    float acc = 0.0f;
#pragma unroll
    for (int i = 0; i < 16; ++i) {
      float4 kv = *(const float4*)(kp + 4 * i);
      acc += qv[i].x * kv.x + qv[i].y * kv.y + qv[i].z * kv.z + qv[i].w * kv.w;
    }
    acc *= 0.125f;                                    // 1/sqrt(64)
    if (mask) {
      float m = mask[((size_t)b * TQ + q) * TK + k];
      if (KNOW) acc *= m;
      if (m == 0.0f) acc = -1.0e9f;
    }
    s[k] = acc;
    mx = fmaxf(mx, acc);
  }
  float sum = 0.0f;
#pragma unroll
  for (int k = 0; k < TK; ++k) { s[k] = __expf(s[k] - mx); sum += s[k]; }
  float inv = 1.0f / sum;

  float* op = O + ((size_t)b * TQ + q) * 512 + h * 64;
  const float* vb = Vm + (size_t)b * TK * 512 + h * 64;
#pragma unroll 2
  for (int d4 = 0; d4 < 16; ++d4) {
    float ox = 0.f, oy = 0.f, oz = 0.f, ow = 0.f;
#pragma unroll
    for (int k = 0; k < TK; ++k) {
      float4 vv = *(const float4*)(vb + (size_t)k * 512 + d4 * 4);
      ox += s[k] * vv.x; oy += s[k] * vv.y; oz += s[k] * vv.z; ow += s[k] * vv.w;
    }
    *(float4*)(op + d4 * 4) = make_float4(ox * inv, oy * inv, oz * inv, ow * inv);
  }
}

// LayerNorm over D=512, Bessel-corrected std, out = g*(x-mean)/(std+eps)+b. One wave per row.
__global__ __launch_bounds__(256) void layernorm_kernel(
    const float* __restrict__ X, const float* __restrict__ g,
    const float* __restrict__ bb, float* __restrict__ Y, int rows)
{
  int row  = blockIdx.x * 8 + (threadIdx.x >> 5);
  int lane = threadIdx.x & 31;
  if (row >= rows) return;
  const float* x = X + (size_t)row * 512;
  float4 v[4]; float sum = 0.0f;
#pragma unroll
  for (int i = 0; i < 4; ++i) {
    v[i] = *(const float4*)(x + lane * 4 + 128 * i);
    sum += v[i].x + v[i].y + v[i].z + v[i].w;
  }
#pragma unroll
  for (int off = 16; off > 0; off >>= 1) sum += __shfl_xor(sum, off, 32);
  float mean = sum * (1.0f / 512.0f);
  float var = 0.0f;
#pragma unroll
  for (int i = 0; i < 4; ++i) {
    float dx = v[i].x - mean, dy = v[i].y - mean, dz = v[i].z - mean, dw = v[i].w - mean;
    var += dx * dx + dy * dy + dz * dz + dw * dw;
  }
#pragma unroll
  for (int off = 16; off > 0; off >>= 1) var += __shfl_xor(var, off, 32);
  float stdv = sqrtf(var * (1.0f / 511.0f));
  float inv  = 1.0f / (stdv + 1e-6f);
  float* y = Y + (size_t)row * 512;
#pragma unroll
  for (int i = 0; i < 4; ++i) {
    int c = lane * 4 + 128 * i;
    float4 gv = *(const float4*)(g + c);
    float4 bv = *(const float4*)(bb + c);
    *(float4*)(y + c) = make_float4(gv.x * (v[i].x - mean) * inv + bv.x,
                                    gv.y * (v[i].y - mean) * inv + bv.y,
                                    gv.z * (v[i].z - mean) * inv + bv.z,
                                    gv.w * (v[i].w - mean) * inv + bv.w);
  }
}

// rel[b,i,j] = dot(er[b,i,:], er[b,j,:]) over 512
__global__ __launch_bounds__(256) void rel_kernel(
    const float* __restrict__ ER, float* __restrict__ REL, int Bn)
{
  int gid = blockIdx.x * blockDim.x + threadIdx.x;
  if (gid >= Bn * 1089) return;
  int j  = gid % 33;
  int i2 = (gid / 33) % 33;
  int b  = gid / 1089;
  const float* a = ER + ((size_t)b * 33 + i2) * 512;
  const float* c = ER + ((size_t)b * 33 + j) * 512;
  float acc = 0.0f;
#pragma unroll 4
  for (int d4 = 0; d4 < 128; ++d4) {
    float4 av = *(const float4*)(a + d4 * 4);
    float4 cv = *(const float4*)(c + d4 * 4);
    acc += av.x * cv.x + av.y * cv.y + av.z * cv.z + av.w * cv.w;
  }
  REL[(size_t)b * 1089 + i2 * 33 + j] = acc;
}

struct WtDesc { const unsigned short* p; int K32; };

extern "C" void kernel_launch(void* const* d_in, const int* in_sizes, int n_in,
                              void* d_out, int out_size, void* d_ws, size_t ws_size,
                              hipStream_t stream)
{
  (void)n_in; (void)out_size; (void)ws_size;
  const float* visual  = (const float*)d_in[0];
  const float* entity  = (const float*)d_in[1];
  const float* graph   = (const float*)d_in[2];
  const float* vis_W   = (const float*)d_in[3];
  const float* vis_b   = (const float*)d_in[4];
  const float* ent_W   = (const float*)d_in[5];
  const float* ent_b   = (const float*)d_in[6];
  const float* ca_Wq   = (const float*)d_in[7];
  const float* ca_bq   = (const float*)d_in[8];
  const float* ca_Wk   = (const float*)d_in[9];
  const float* ca_bk   = (const float*)d_in[10];
  const float* ca_Wv   = (const float*)d_in[11];
  const float* ca_bv   = (const float*)d_in[12];
  const float* ca_Wo   = (const float*)d_in[13];
  const float* ca_bo   = (const float*)d_in[14];
  const float* kp_W    = (const float*)d_in[15];
  const float* kp_b    = (const float*)d_in[16];
  const float* pred_W  = (const float*)d_in[17];
  const float* pred_b  = (const float*)d_in[18];
  const float* l_Wq    = (const float*)d_in[19];
  const float* l_bq    = (const float*)d_in[20];
  const float* l_Wk    = (const float*)d_in[21];
  const float* l_bk    = (const float*)d_in[22];
  const float* l_Wv    = (const float*)d_in[23];
  const float* l_bv    = (const float*)d_in[24];
  const float* l_Wo    = (const float*)d_in[25];
  const float* l_bo    = (const float*)d_in[26];
  const float* l_ffW1  = (const float*)d_in[27];
  const float* l_ffb1  = (const float*)d_in[28];
  const float* l_ffW2  = (const float*)d_in[29];
  const float* l_ffb2  = (const float*)d_in[30];
  const float* l_n1_g  = (const float*)d_in[31];
  const float* l_n1_b  = (const float*)d_in[32];
  const float* l_n2_g  = (const float*)d_in[33];
  const float* l_n2_b  = (const float*)d_in[34];
  const float* fin_g   = (const float*)d_in[35];
  const float* fin_b   = (const float*)d_in[36];
  const float* ad_W1   = (const float*)d_in[37];
  const float* ad_b1   = (const float*)d_in[38];
  const float* ad_W2   = (const float*)d_in[39];
  const float* ad_b2   = (const float*)d_in[40];
  const float* sc_fc_W = (const float*)d_in[41];
  const float* sc_fc_b = (const float*)d_in[42];
  const float* sc_cls_W= (const float*)d_in[43];
  const float* sc_cls_b= (const float*)d_in[44];

  const int Bn  = in_sizes[0] / (24 * 2048);   // 512
  const int M33 = Bn * 33;
  const int M24 = Bn * 24;

  // workspace layout
  float* ws = (float*)d_ws;
  const size_t R33 = (size_t)M33 * 512;
  const size_t R24 = (size_t)M24 * 512;
  float* E   = ws;                 // entity proj
  float* V   = E   + R33;          // visual proj
  float* Q   = V   + R24;          // q / er / ff1 / ad1 (reused)
  float* K2  = Q   + R33;
  float* V2  = K2  + R33;
  float* O   = V2  + R33;          // attention output (head-concat)
  float* X   = O   + R33;          // residual stream (ef -> x)
  float* XN  = X   + R33;          // layernorm output
  float* REL = XN  + R33;          // Bn*1089
  float* ADJ = REL + (size_t)Bn * 1089;
  float* CIN = ADJ + (size_t)Bn * 1089;  // Bn*4224
  unsigned short* wpool = (unsigned short*)(CIN + (size_t)Bn * 4224);

  float* adapted = (float*)d_out;                      // (Bn*33, 4096)
  float* cls     = adapted + (size_t)M33 * 4096;       // (Bn, 32)

  // pre-convert all weights: fp32 [K][N] -> bf16 transposed padded [Npad][K32]
  unsigned short* wp = wpool;
  auto prep = [&](const float* W, int K, int N) -> WtDesc {
    int K32  = (K + 31) & ~31;
    int Npad = ((N + BN - 1) / BN) * BN;
    unsigned short* dst = wp;
    size_t total = (size_t)K32 * Npad;
    wp += total;
    convert_wt_kernel<<<(int)((total + 255) / 256), 256, 0, stream>>>(W, dst, K, N, K32, Npad);
    WtDesc d; d.p = dst; d.K32 = K32;
    return d;
  };

  WtDesc w_vis  = prep(vis_W, 2048, 512);
  WtDesc w_ent  = prep(ent_W, 4096, 512);
  WtDesc w_caq  = prep(ca_Wq, 512, 512);
  WtDesc w_cak  = prep(ca_Wk, 512, 512);
  WtDesc w_cav  = prep(ca_Wv, 512, 512);
  WtDesc w_cao  = prep(ca_Wo, 512, 512);
  WtDesc w_kp   = prep(kp_W,  512, 512);
  WtDesc w_pred = prep(pred_W, 1089, 1089);
  WtDesc w_lq[3], w_lk[3], w_lv[3], w_lo[3], w_f1[3], w_f2[3];
  for (int i = 0; i < 3; ++i) {
    const size_t wOff = (size_t)i * 512 * 512;
    w_lq[i] = prep(l_Wq  + wOff, 512, 512);
    w_lk[i] = prep(l_Wk  + wOff, 512, 512);
    w_lv[i] = prep(l_Wv  + wOff, 512, 512);
    w_lo[i] = prep(l_Wo  + wOff, 512, 512);
    w_f1[i] = prep(l_ffW1 + wOff, 512, 512);
    w_f2[i] = prep(l_ffW2 + wOff, 512, 512);
  }
  WtDesc w_ad1 = prep(ad_W1, 512, 512);
  WtDesc w_ad2 = prep(ad_W2, 512, 4096);
  WtDesc w_fc  = prep(sc_fc_W, 512, 128);
  WtDesc w_cls = prep(sc_cls_W, 4224, 32);

  auto gemm = [&](const float* Aa, WtDesc W, const float* bb,
                  const float* rr, float* Cc, int M, int N, int K,
                  float alpha, float beta, int epi) {
    dim3 grid((N + BN - 1) / BN, (M + BM - 1) / BM);
    gemm_wmma_kernel<<<grid, 256, 0, stream>>>(Aa, W.p, bb, rr, Cc, M, N, K, W.K32,
                                               alpha, beta, epi);
  };

  // ---- input projections ----
  gemm(visual, w_vis, vis_b, nullptr, V, M24, 512, 2048, 1.f, 0.f, 0);
  gemm(entity, w_ent, ent_b, nullptr, E, M33, 512, 4096, 1.f, 0.f, 0);

  // ---- cross attention: q from entities (33), kv from visual (24) ----
  gemm(E, w_caq, ca_bq, nullptr, Q,  M33, 512, 512, 1.f, 0.f, 0);
  gemm(V, w_cak, ca_bk, nullptr, K2, M24, 512, 512, 1.f, 0.f, 0);
  gemm(V, w_cav, ca_bv, nullptr, V2, M24, 512, 512, 1.f, 0.f, 0);
  {
    int total = Bn * 8 * 33;
    attn_kernel<33, 24, false><<<(total + 255) / 256, 256, 0, stream>>>(Q, K2, V2, O, nullptr, Bn);
  }
  gemm(O, w_cao, ca_bo, nullptr, X, M33, 512, 512, 1.f, 0.f, 0);   // ef

  // ---- adjacency branch ----
  gemm(X, w_kp, kp_b, nullptr, Q, M33, 512, 512, 1.f, 0.f, 0);     // er
  {
    int total = Bn * 1089;
    rel_kernel<<<(total + 255) / 256, 256, 0, stream>>>(Q, REL, Bn);
  }
  // adj = 0.9*graph + 0.1*sigmoid(rel @ pred_W + pred_b)
  gemm(REL, w_pred, pred_b, graph, ADJ, Bn, 1089, 1089, 0.1f, 0.9f, 2);

  // ---- 3 transformer layers (pre-norm on q only, masked attention, know=True) ----
  for (int i = 0; i < 3; ++i) {
    const size_t bOff = (size_t)i * 512;
    layernorm_kernel<<<(M33 + 7) / 8, 256, 0, stream>>>(X, l_n1_g + bOff, l_n1_b + bOff, XN, M33);
    gemm(XN, w_lq[i], l_bq + bOff, nullptr, Q,  M33, 512, 512, 1.f, 0.f, 0);
    gemm(X,  w_lk[i], l_bk + bOff, nullptr, K2, M33, 512, 512, 1.f, 0.f, 0);
    gemm(X,  w_lv[i], l_bv + bOff, nullptr, V2, M33, 512, 512, 1.f, 0.f, 0);
    {
      int total = Bn * 8 * 33;
      attn_kernel<33, 33, true><<<(total + 255) / 256, 256, 0, stream>>>(Q, K2, V2, O, ADJ, Bn);
    }
    gemm(O, w_lo[i], l_bo + bOff, X, X, M33, 512, 512, 1.f, 1.f, 0);   // x += att
    layernorm_kernel<<<(M33 + 7) / 8, 256, 0, stream>>>(X, l_n2_g + bOff, l_n2_b + bOff, XN, M33);
    gemm(XN, w_f1[i], l_ffb1 + bOff, nullptr, Q, M33, 512, 512, 1.f, 0.f, 1);  // relu
    gemm(Q,  w_f2[i], l_ffb2 + bOff, X, X, M33, 512, 512, 1.f, 1.f, 0);        // x += ff
  }

  // ---- heads ----
  layernorm_kernel<<<(M33 + 7) / 8, 256, 0, stream>>>(X, fin_g, fin_b, XN, M33);  // emb
  gemm(XN, w_ad1, ad_b1, nullptr, Q, M33, 512, 512, 1.f, 0.f, 0);
  gemm(Q, w_ad2, ad_b2, nullptr, adapted, M33, 4096, 512, 1.f, 0.f, 0);
  gemm(XN, w_fc, sc_fc_b, nullptr, CIN, M33, 128, 512, 1.f, 0.f, 0);
  gemm(CIN, w_cls, sc_cls_b, nullptr, cls, Bn, 32, 4224, 1.f, 0.f, 2);
}
